// RNN_seplay_blankII_86148454023840
// MI455X (gfx1250) — matrix operand order repaired and checked by reading, not compiled, not run on hardware
//
#include <hip/hip_runtime.h>
#include <hip/hip_bf16.h>
#include <stddef.h>

// ---------------------------------------------------------------------------
// RNN with separate E/I populations, blank (1-I) inhibitory recurrence.
// Sequential scan over T-1=2047 steps; batch (32) split as 2 workgroups x 16.
// Per step: 3x [16x256]@[256x256] matmuls via v_wmma_f32_16x16x32_f16, with
// the input projections (s_in @ W_ine.T / W_ini.T) folded in as a 9th K-tile.
// B-fragment loads are written in-loop with loop-invariant addresses so the
// compiler's pressure-aware LICM keeps as many as fit register-resident
// (extended VGPR file) and streams the remainder. Next step's noise rows are
// prefetched during the WMMA phase to hide HBM latency.
// ---------------------------------------------------------------------------

typedef __attribute__((ext_vector_type(16))) _Float16 v16h;
typedef __attribute__((ext_vector_type(8)))  _Float16 v8h;
typedef __attribute__((ext_vector_type(8)))  float    v8f;

#define NEC   256
#define TLEN  2048
#define BATCH 32
#define AROW  296              // padded f16 row stride: 256 data + 32 ext + pad
#define PACK_MAIN  (3 * 16 * 8 * 512)   // 196608 halfs: W_ee, -W_ie, W_ei
#define PACK_EXT   (2 * 16 * 512)       // 16384 halfs: W_ine, W_ini (K-padded)

__device__ __forceinline__ v8f wmma_f16(v16h a, v16h b, v8f c) {
    // (neg_a, A, neg_b, B, c_mod, C, reuse_a, reuse_b)
    return __builtin_amdgcn_wmma_f32_16x16x32_f16(false, a, false, b,
                                                  (short)0, c, false, false);
}

// A fragment (16x32 f16, K-tile kt) from LDS row-major [m][k] with padding.
// Lane L: m = L&15; element i -> k = kt*32 + 8*(L>>4) + (i&7) + 16*(i>>3).
__device__ __forceinline__ v16h loadA(const _Float16* rowBase, int kt) {
    const v8h lo = *(const v8h*)(rowBase + kt * 32);
    const v8h hi = *(const v8h*)(rowBase + kt * 32 + 16);
    return __builtin_shufflevector(lo, hi, 0,1,2,3,4,5,6,7,8,9,10,11,12,13,14,15);
}

// B fragment: one contiguous 32-byte load per lane; baseByte is per-lane and
// loop-invariant (compiler hoists what fits into persistent registers).
__device__ __forceinline__ v16h loadBo(const char* packB, int baseByte, int kt) {
    return *(const v16h*)(packB + baseByte + kt * 1024);
}

// ---------------------------------------------------------------------------
// Prepack: B[k][n] = W[n*256+k] (so D = A x B computes A @ W.T), f16,
// stored in exact per-lane fragment order; mIdx==1 (W_ie) is negated.
// Fragment element i at lane L: n = ntile*16 + (L&15), k = kt*32+8*(L>>4)+(i&7)+16*(i>>3).
// Ext region (after PACK_MAIN): W_ine (mat 0), W_ini (mat 1), single K-tile,
// zero beyond k'=7.
// ---------------------------------------------------------------------------
__global__ __launch_bounds__(256)
void prepack_weights(const float* __restrict__ W_ee,
                     const float* __restrict__ W_ie,
                     const float* __restrict__ W_ei,
                     const float* __restrict__ W_ine,
                     const float* __restrict__ W_ini,
                     _Float16* __restrict__ pack) {
    int gid = blockIdx.x * 256 + threadIdx.x;   // PACK_MAIN + PACK_EXT total
    if (gid < PACK_MAIN) {
        int i     =  gid        & 15;
        int lane  = (gid >> 4)  & 31;
        int kt    = (gid >> 9)  & 7;
        int ntile = (gid >> 12) & 15;
        int mIdx  =  gid >> 16;                 // 0..2
        int n = ntile * 16 + (lane & 15);
        int k = kt * 32 + ((lane >> 4) << 3) + (i & 7) + ((i >> 3) << 4);
        const float* W = (mIdx == 0) ? W_ee : ((mIdx == 1) ? W_ie : W_ei);
        float v = W[n * NEC + k];
        if (mIdx == 1) v = -v;
        pack[gid] = (_Float16)v;
    } else {
        int g2    = gid - PACK_MAIN;            // 0..16383
        int i     =  g2        & 15;
        int lane  = (g2 >> 4)  & 31;
        int ntile = (g2 >> 9)  & 15;
        int mat   =  g2 >> 13;                  // 0: W_ine, 1: W_ini
        int n  = ntile * 16 + (lane & 15);
        int kp = ((lane >> 4) << 3) + (i & 7) + ((i >> 3) << 4);        // 0..31
        const float* W = mat ? W_ini : W_ine;
        float v = (kp < 8) ? W[n * 8 + kp] : 0.f;
        pack[gid] = (_Float16)v;
    }
}

// ---------------------------------------------------------------------------
// Main recurrent kernel: grid = 2 blocks x 256 threads (8 waves).
// Block handles batches [blockIdx*16, +16).
// ---------------------------------------------------------------------------
__global__ __launch_bounds__(256)
void rnn_scan(const float* __restrict__ u,
              const float* __restrict__ init_in,
              const float* __restrict__ init_e,
              const float* __restrict__ init_i,
              const float* __restrict__ bias_corr,
              const float* __restrict__ noise_e,
              const float* __restrict__ noise_i,
              const float* __restrict__ W_out,
              const float* __restrict__ wii0_p,
              const _Float16* __restrict__ pack,
              float* __restrict__ o_out,
              float* __restrict__ o_stin,
              float* __restrict__ o_acte,
              float* __restrict__ o_acti,
              float* __restrict__ o_ste,
              float* __restrict__ o_sti) {
    __shared__ float    sE[16 * NEC];        // 16 KB  f32 e-state
    __shared__ float    sI[16 * NEC];        // 16 KB  f32 i-state
    __shared__ _Float16 reE[16 * AROW];      // relu(e) f16 + s_in ext (A op)
    __shared__ _Float16 riI[16 * AROW];      // relu(i) f16 (A operand)
    __shared__ float    sIn[16 * 8];
    __shared__ float    bcS[16 * 8];
    __shared__ float    Wout[2 * NEC];       // 2 KB
    __shared__ float    rowsum[16];
    __shared__ float    partialOut[256 * 2]; // 2 KB

    const int tid   = threadIdx.x;
    const int lane  = tid & 31;
    const int wave  = tid >> 5;
    const int bBase = blockIdx.x * 16;
    const float a = 0.1f, oma = 0.9f;
    const float ns = 0.022360679774997896f;  // sqrt(2*alpha*sigma^2)
    const float wii0 = *wii0_p;

    // ---- prologue: stage constants + init state, emit t=0 outputs ----
    for (int idx = tid; idx < 2 * NEC; idx += 256) Wout[idx] = W_out[idx];
    // zero the A-row extension regions (k' >= 8 of reE ext; all of riI ext)
    for (int idx = tid; idx < 16 * 32; idx += 256) {
        int m = idx >> 5, x = idx & 31;
        riI[m * AROW + 256 + x] = (_Float16)0.f;
        if (x >= 8) reE[m * AROW + 256 + x] = (_Float16)0.f;
    }
    if (tid < 128) {
        int m = tid >> 3, d = tid & 7;
        bcS[tid] = bias_corr[(size_t)(bBase + m) * 8 + d];
        float v = init_in[(size_t)(bBase + m) * 8 + d];
        sIn[tid] = v;
        reE[m * AROW + 256 + d] = (_Float16)v;     // s_in ext for K-tile 8
        o_stin[((size_t)(bBase + m) * TLEN + 0) * 8 + d] = v;
    }
    for (int idx = tid; idx < 16 * NEC; idx += 256) {
        int m = idx >> 8, j = idx & 255;
        float e0 = init_e[(size_t)(bBase + m) * NEC + j];
        float i0 = init_i[(size_t)(bBase + m) * NEC + j];
        sE[idx] = e0; sI[idx] = i0;
        float ae = fmaxf(e0, 0.f), ai = fmaxf(i0, 0.f);
        reE[m * AROW + j] = (_Float16)ae;
        riI[m * AROW + j] = (_Float16)ai;
        size_t g = ((size_t)(bBase + m) * TLEN + 0) * NEC + j;
        o_ste[g] = e0; o_sti[g] = i0; o_acte[g] = ae; o_acti[g] = ai;
    }
    if (tid < 16) rowsum[tid] = 0.f;
    __syncthreads();
    {   // partials for out[t=0] (4-way split accumulation)
        int m = tid >> 4, seg = tid & 15;
        float s0a = 0.f, s0b = 0.f, s1a = 0.f, s1b = 0.f;
        for (int jj = 0; jj < 16; jj += 2) {
            int j0 = seg * 16 + jj, j1 = j0 + 1;
            float a0 = fmaxf(sE[m * NEC + j0], 0.f);
            float a1 = fmaxf(sE[m * NEC + j1], 0.f);
            s0a += a0 * Wout[j0];       s0b += a1 * Wout[j1];
            s1a += a0 * Wout[NEC + j0]; s1b += a1 * Wout[NEC + j1];
        }
        partialOut[tid * 2 + 0] = s0a + s0b;
        partialOut[tid * 2 + 1] = s1a + s1b;
    }
    __syncthreads();

    const int mrow  = lane & 15;
    const int khalf = lane >> 4;
    const _Float16* reRow = &reE[mrow * AROW + khalf * 8];
    const _Float16* riRow = &riI[mrow * AROW + khalf * 8];
    const int n0 = wave * 2;   // this wave's N-tile pair

    // Per-lane, loop-invariant byte offsets into the packed B streams.
    const char* packB = (const char*)pack;
    const int lb = lane * 32;
    const int bE0 = ((0 * 16 + n0    ) * 256) * 32 + lb;  // W_ee,  tile n0
    const int bE1 = ((0 * 16 + n0 + 1) * 256) * 32 + lb;  // W_ee,  tile n0+1
    const int bN0 = ((1 * 16 + n0    ) * 256) * 32 + lb;  // -W_ie, tile n0
    const int bN1 = ((1 * 16 + n0 + 1) * 256) * 32 + lb;  // -W_ie, tile n0+1
    const int bI0 = ((2 * 16 + n0    ) * 256) * 32 + lb;  // W_ei,  tile n0
    const int bI1 = ((2 * 16 + n0 + 1) * 256) * 32 + lb;  // W_ei,  tile n0+1
    const int xE0 = PACK_MAIN * 2 + ((0 * 16 + n0    ) * 512 + lane * 16) * 2;
    const int xE1 = PACK_MAIN * 2 + ((0 * 16 + n0 + 1) * 512 + lane * 16) * 2;
    const int xI0 = PACK_MAIN * 2 + ((1 * 16 + n0    ) * 512 + lane * 16) * 2;
    const int xI1 = PACK_MAIN * 2 + ((1 * 16 + n0 + 1) * 512 + lane * 16) * 2;

    // ---- sequential scan ----
    for (int t = 0; t < TLEN - 1; ++t) {
        // ---- phase A: out[t] store, noise prefetch (t+1), rowsum(ri),
        //      WMMA accumulation ----
        if (tid < 32) {
            int m = tid >> 1, o = tid & 1;
            float pa = 0.f, pb = 0.f, pc = 0.f, pd = 0.f;
            for (int seg = 0; seg < 16; seg += 4) {
                pa += partialOut[(m * 16 + seg + 0) * 2 + o];
                pb += partialOut[(m * 16 + seg + 1) * 2 + o];
                pc += partialOut[(m * 16 + seg + 2) * 2 + o];
                pd += partialOut[(m * 16 + seg + 3) * 2 + o];
            }
            o_out[((size_t)(bBase + m) * TLEN + t) * 2 + o] = (pa + pb) + (pc + pd);
        }
        {   // prefetch next step's noise rows (hidden behind the WMMA phase)
            int pm = tid >> 4, seg = tid & 15;
            size_t nbase = ((size_t)(bBase + pm) * TLEN + (t + 1)) * NEC + seg * 16;
            __builtin_prefetch(&noise_e[nbase], 0, 1);
            __builtin_prefetch(&noise_i[nbase], 0, 1);
        }
        {   // rowsum of relu(s_i), 4-way split accumulation + LDS atomic
            int m = tid >> 4, seg = tid & 15;
            float pa = 0.f, pb = 0.f, pc = 0.f, pd = 0.f;
            const float* r = &sI[m * NEC + seg * 16];
            for (int jj = 0; jj < 16; jj += 4) {
                pa += fmaxf(r[jj + 0], 0.f);
                pb += fmaxf(r[jj + 1], 0.f);
                pc += fmaxf(r[jj + 2], 0.f);
                pd += fmaxf(r[jj + 3], 0.f);
            }
            atomicAdd(&rowsum[m], (pa + pb) + (pc + pd));
        }
        v8f accE0 = {}, accE1 = {}, accI0 = {}, accI1 = {};
        #pragma unroll
        for (int kt = 0; kt < 8; ++kt) {
            v16h aE = loadA(reRow, kt);
            v16h aI = loadA(riRow, kt);
            v16h b0 = loadBo(packB, bE0, kt);
            v16h b1 = loadBo(packB, bE1, kt);
            v16h b2 = loadBo(packB, bN0, kt);
            v16h b3 = loadBo(packB, bN1, kt);
            v16h b4 = loadBo(packB, bI0, kt);
            v16h b5 = loadBo(packB, bI1, kt);
            accE0 = wmma_f16(aE, b0, accE0);
            accE1 = wmma_f16(aE, b1, accE1);
            accI0 = wmma_f16(aE, b4, accI0);
            accI1 = wmma_f16(aE, b5, accI1);
            accE0 = wmma_f16(aI, b2, accE0);   // -W_ie contribution
            accE1 = wmma_f16(aI, b3, accE1);
        }
        {   // K-tile 8: input projection (A = s_in ext, B = W_ine/W_ini)
            v16h aX = loadA(reRow, 8);
            v16h c0 = loadBo(packB, xE0, 0);
            v16h c1 = loadBo(packB, xE1, 0);
            v16h c2 = loadBo(packB, xI0, 0);
            v16h c3 = loadBo(packB, xI1, 0);
            accE0 = wmma_f16(aX, c0, accE0);
            accE1 = wmma_f16(aX, c1, accE1);
            accI0 = wmma_f16(aX, c2, accI0);
            accI1 = wmma_f16(aX, c3, accI1);
        }
        __syncthreads();

        // ---- phase B: per-element state update (C layout: m=v+8*(lane>>4),
        //      n=lane&15) ----
        {
            const int mh = lane >> 4;
            const int nl = lane & 15;
            #pragma unroll
            for (int tile = 0; tile < 2; ++tile) {
                const int j = (n0 + tile) * 16 + nl;
                #pragma unroll
                for (int v = 0; v < 8; ++v) {
                    int m  = v + 8 * mh;
                    int bg = bBase + m;
                    float me   = (tile ? accE1[v] : accE0[v]);
                    float olde = sE[m * NEC + j];
                    float e_new = oma * olde + a * me +
                        ns * noise_e[((size_t)bg * TLEN + t) * NEC + j];
                    float ae = fmaxf(e_new, 0.f);
                    sE[m * NEC + j]   = e_new;
                    reE[m * AROW + j] = (_Float16)ae;
                    size_t g = ((size_t)bg * TLEN + (t + 1)) * NEC + j;
                    o_ste[g]  = e_new;
                    o_acte[g] = ae;

                    float oldi   = sI[m * NEC + j];
                    float ri_old = fmaxf(oldi, 0.f);
                    float mi = (tile ? accI1[v] : accI0[v])
                             - wii0 * (rowsum[m] - ri_old);
                    float i_new = oma * oldi + a * mi +
                        ns * noise_i[((size_t)bg * TLEN + t) * NEC + j];
                    float ai = fmaxf(i_new, 0.f);
                    sI[m * NEC + j]   = i_new;
                    riI[m * AROW + j] = (_Float16)ai;
                    o_sti[g]  = i_new;
                    o_acti[g] = ai;
                }
            }
        }
        __syncthreads();

        // ---- phase C: W_out partials (t+1), s_in update, rowsum clear ----
        {
            int m = tid >> 4, seg = tid & 15;
            float s0a = 0.f, s0b = 0.f, s1a = 0.f, s1b = 0.f;
            for (int jj = 0; jj < 16; jj += 2) {
                int j0 = seg * 16 + jj, j1 = j0 + 1;
                float a0 = fmaxf(sE[m * NEC + j0], 0.f);
                float a1 = fmaxf(sE[m * NEC + j1], 0.f);
                s0a += a0 * Wout[j0];       s0b += a1 * Wout[j1];
                s1a += a0 * Wout[NEC + j0]; s1b += a1 * Wout[NEC + j1];
            }
            partialOut[tid * 2 + 0] = s0a + s0b;
            partialOut[tid * 2 + 1] = s1a + s1b;
        }
        if (tid < 128) {
            int m = tid >> 3, d = tid & 7;
            int bg = bBase + m;
            float inNew = oma * sIn[tid] +
                a * (u[((size_t)bg * TLEN + t) * 8 + d] + bcS[tid]);
            sIn[tid] = inNew;
            reE[m * AROW + 256 + d] = (_Float16)inNew;   // keep ext in sync
            o_stin[((size_t)bg * TLEN + (t + 1)) * 8 + d] = inNew;
            if (t + 1 < TLEN - 1)   // prefetch next u row
                __builtin_prefetch(&u[((size_t)bg * TLEN + (t + 1)) * 8 + d], 0, 1);
        }
        if (tid < 16) rowsum[tid] = 0.f;
        __syncthreads();
    }

    // ---- epilogue: out[T-1] ----
    if (tid < 32) {
        int m = tid >> 1, o = tid & 1;
        float pa = 0.f, pb = 0.f, pc = 0.f, pd = 0.f;
        for (int seg = 0; seg < 16; seg += 4) {
            pa += partialOut[(m * 16 + seg + 0) * 2 + o];
            pb += partialOut[(m * 16 + seg + 1) * 2 + o];
            pc += partialOut[(m * 16 + seg + 2) * 2 + o];
            pd += partialOut[(m * 16 + seg + 3) * 2 + o];
        }
        o_out[((size_t)(bBase + m) * TLEN + (TLEN - 1)) * 2 + o] = (pa + pb) + (pc + pd);
    }
}

extern "C" void kernel_launch(void* const* d_in, const int* in_sizes, int n_in,
                              void* d_out, int out_size, void* d_ws, size_t ws_size,
                              hipStream_t stream) {
    (void)in_sizes; (void)n_in; (void)out_size; (void)ws_size;
    // setup_inputs order:
    const float* u         = (const float*)d_in[0];
    const float* init_in   = (const float*)d_in[1];
    const float* init_e    = (const float*)d_in[2];
    const float* init_i    = (const float*)d_in[3];
    const float* bias_corr = (const float*)d_in[4];
    const float* noise_e   = (const float*)d_in[5];
    const float* noise_i   = (const float*)d_in[6];
    const float* W_ee      = (const float*)d_in[7];
    const float* W_ei      = (const float*)d_in[8];   // note: W_ei before W_ie
    const float* W_ie      = (const float*)d_in[9];
    const float* W_ine     = (const float*)d_in[10];
    const float* W_ini     = (const float*)d_in[11];
    const float* W_out     = (const float*)d_in[12];
    const float* wii0      = (const float*)d_in[13];

    _Float16* pack = (_Float16*)d_ws;                 // ~416 KB of scratch

    // Outputs concatenated flat in return order.
    float* o_out  = (float*)d_out;                    // (B,T,2)
    float* o_stin = o_out  + (size_t)BATCH * TLEN * 2;
    float* o_acte = o_stin + (size_t)BATCH * TLEN * 8;
    float* o_acti = o_acte + (size_t)BATCH * TLEN * NEC;
    float* o_ste  = o_acti + (size_t)BATCH * TLEN * NEC;
    float* o_sti  = o_ste  + (size_t)BATCH * TLEN * NEC;

    const int packElems  = PACK_MAIN + PACK_EXT;      // 212992
    prepack_weights<<<packElems / 256, 256, 0, stream>>>(W_ee, W_ie, W_ei,
                                                         W_ine, W_ini, pack);
    rnn_scan<<<2, 256, 0, stream>>>(u, init_in, init_e, init_i, bias_corr,
                                    noise_e, noise_i, W_out,
                                    wii0, pack,
                                    o_out, o_stin, o_acte, o_acti, o_ste, o_sti);
}
